// ColorStableBilateralFilter_86414741995917
// MI455X (gfx1250) — matrix operand order repaired and checked by reading, not compiled, not run on hardware
//
#include <hip/hip_runtime.h>
#include <hip/hip_bf16.h>

typedef __attribute__((ext_vector_type(16))) _Float16 v16h;
typedef __attribute__((ext_vector_type(8)))  _Float16 v8h;
typedef __attribute__((ext_vector_type(8)))  float    v8f;
typedef unsigned int uint4v __attribute__((ext_vector_type(4)));
typedef int          int8v  __attribute__((ext_vector_type(8)));
typedef int          int4v  __attribute__((ext_vector_type(4)));

// ---------------------------------------------------------------------------
// Weight packing: fp32 OIHW (Cout,Cin,3,3) -> per-lane A-fragment f16 layout
// for V_WMMA_F32_16X16X32_F16 (A = 16 couts x 32 K, K = cin x taps).
//   lanes 0-15 : v0..v7 -> K = 2v,2v+1
//   lanes16-31 : +16
// Cin=32 -> chunk = 1 tap (9 chunks); Cin=16 -> chunk = 2 taps (5 chunks,
// last half zero).  Memory layout: [cout_tile][chunk][lane][16] f16.
// ---------------------------------------------------------------------------
__device__ __forceinline__ int kmapA(int v, int half, int e) {
  int base = 2 * v + 16 * half;
  return base + e;
}

__global__ void pack_w_kernel(const float* __restrict__ w, _Float16* __restrict__ wp,
                              int CIN) {
  const int lane  = threadIdx.x;          // block = 32
  const int nch   = (CIN == 32) ? 9 : 5;
  const int chunk = blockIdx.x % nch;
  const int ct    = blockIdx.x / nch;
  const int m     = ct * 16 + (lane & 15);
  const int half  = lane >> 4;
  v16h ov;
  #pragma unroll
  for (int v = 0; v < 8; ++v) {
    #pragma unroll
    for (int e = 0; e < 2; ++e) {
      int k = kmapA(v, half, e);
      int tap, cin;
      if (CIN == 32) { tap = chunk; cin = k; }
      else           { tap = chunk * 2 + (k >> 4); cin = k & 15; }
      float val = 0.f;
      if (tap < 9) val = w[(m * CIN + cin) * 9 + tap];
      ov[v * 2 + e] = (_Float16)val;
    }
  }
  *(v16h*)(wp + ((size_t)blockIdx.x * 32 + lane) * 16) = ov;
}

// ---------------------------------------------------------------------------
// x (fp32 NCHW, 16ch) -> padded NHWC f16 [B][H+2][W+2][16]
// ---------------------------------------------------------------------------
__global__ void x_to_nhwc_kernel(const float* __restrict__ x, _Float16* __restrict__ o) {
  int i = blockIdx.x * 256 + threadIdx.x;
  const int total = 4 * 512 * 512 * 16;
  if (i >= total) return;
  int c = i & 15;
  int t = i >> 4;
  int w = t % 512;
  int h = (t / 512) % 512;
  int b = t / (512 * 512);
  o[(((size_t)b * 514 + h + 1) * 514 + w + 1) * 16 + c] =
      (_Float16)x[(((size_t)b * 16 + c) * 512 + h) * 512 + w];
}

// Zero the 1-pixel halo ring of a padded NHWC f16 buffer [B][H+2][W+2][C].
__global__ void zero_border_kernel(_Float16* __restrict__ buf, int B, int H, int W, int C) {
  const int Hp = H + 2, Wp = W + 2;
  const int nb = 2 * Wp + 2 * H;            // border pixels per image
  const int total = B * nb * C;
  int i = blockIdx.x * 256 + threadIdx.x;
  if (i >= total) return;
  int c = i % C;
  int t = i / C;
  int pix = t % nb;
  int b = t / nb;
  int h, w;
  if (pix < Wp)            { h = 0;      w = pix; }
  else if (pix < 2 * Wp)   { h = Hp - 1; w = pix - Wp; }
  else { int q = pix - 2 * Wp; h = 1 + (q >> 1); w = (q & 1) ? (Wp - 1) : 0; }
  buf[(((size_t)b * Hp + h) * Wp + w) * C + c] = (_Float16)0.f;
}

// ---------------------------------------------------------------------------
// 3x3 SAME conv as implicit GEMM on the WMMA pipe, f16 NHWC in/out (padded).
// Block = 256 threads (8 wave32); one output row x 128 px x one 16-cout tile.
// Input tile (3 rows x 130 px x CIN f16) is fetched by ONE Tensor-Data-Mover
// descriptor (2D: dim0 = 130*CIN contiguous f16, dim1 = 3 rows, stride =
// (W+2)*CIN) issued by wave 0, waited with s_wait_tensorcnt, then barrier.
// LDS layout [row][px][cin]: each lane's B column is one contiguous 32B load.
// ---------------------------------------------------------------------------
template <int CIN, int COUT, bool RELU>
__global__ __launch_bounds__(256)
void conv3x3_wmma_kernel(const _Float16* __restrict__ in,
                         const _Float16* __restrict__ wp,
                         const float* __restrict__ bias,
                         _Float16* __restrict__ out, int H, int W) {
  constexpr int NCH = (CIN == 32) ? 9 : 5;
  constexpr int NT  = COUT / 16;
  constexpr int PW  = 130;
  __shared__ alignas(32) _Float16 lds[3 * PW * CIN];

  const int tid  = threadIdx.x;
  const int lane = tid & 31;
  const int wav  = tid >> 5;
  const int w0   = blockIdx.x * 128;
  const int h    = blockIdx.y;
  const int b    = blockIdx.z / NT;
  const int ct   = blockIdx.z % NT;
  const int Wp   = W + 2, Hp = H + 2;

  // ---- TDM async tile load (one descriptor per block, issued by wave 0) ----
  if (wav == 0) {
    const unsigned rowElems = (unsigned)(PW * CIN);                 // f16 per row
    const unsigned long long gaddr = (unsigned long long)(uintptr_t)
        (in + (((size_t)b * Hp + h) * Wp + w0) * CIN);
    const unsigned long long stride0 = (unsigned long long)Wp * CIN; // elements
    const unsigned ldsOff = (unsigned)(uintptr_t)(void*)&lds[0];

    uint4v g0;
    g0[0] = 1u;                                           // count=1, user mode
    g0[1] = ldsOff;                                       // lds_addr
    g0[2] = (unsigned)(gaddr & 0xFFFFFFFFu);              // global_addr lo
    g0[3] = (unsigned)((gaddr >> 32) & 0x1FFFFFFu) | (2u << 30); // hi | type=2

    int8v g1;
    g1[0] = (int)(1u << 16);                              // data_size=1 (2B)
    g1[1] = (int)((rowElems & 0xFFFFu) << 16);            // tensor_dim0[15:0]
    g1[2] = (int)(((rowElems >> 16) & 0xFFFFu) | (3u << 16)); // td0 hi | td1 lo (=3)
    g1[3] = (int)((rowElems & 0xFFFFu) << 16);            // td1 hi(0) | tile_dim0
    g1[4] = (int)3u;                                      // tile_dim1=3, tile_dim2=0
    g1[5] = (int)(stride0 & 0xFFFFFFFFu);                 // dim0_stride lo32
    g1[6] = (int)((stride0 >> 32) & 0xFFFFu);             // dim0_stride hi16
    g1[7] = 0;

    int4v zz4 = {};
    int8v zz8 = {};
    __builtin_amdgcn_tensor_load_to_lds(g0, g1, zz4, zz4, zz8, 0);
    __builtin_amdgcn_s_wait_tensorcnt(0);
  }
  __syncthreads();

  const int n    = lane & 15;
  const int half = lane >> 4;
  const int lp   = wav * 16 + n;     // local pixel; LDS px = lp + dx (halo=1)

  v8f acc = {};
  const _Float16* wpt = wp + (size_t)ct * NCH * 32 * 16;

  #pragma unroll
  for (int chunk = 0; chunk < NCH; ++chunk) {
    v16h a = *(const v16h*)(wpt + ((size_t)chunk * 32 + lane) * 16);
    v16h bf = {};
    if constexpr (CIN == 32) {
      const int dy = chunk / 3, dx = chunk % 3;
      bf = *(const v16h*)&lds[((dy * PW) + lp + dx) * CIN + half * 16];
    } else {
      const int tap = chunk * 2 + half;   // lane half selects the sub-tap
      if (tap < 9) {
        const int dy = tap / 3, dx = tap % 3;
        bf = *(const v16h*)&lds[((dy * PW) + lp + dx) * CIN];
      }
    }
    acc = __builtin_amdgcn_wmma_f32_16x16x32_f16(
        false, a, false, bf, (short)0, acc, false, false);
  }

  // C/D layout: VGPR r -> M = r + 8*half (contiguous couts per lane!),
  // N = lane&15 -> pixel.  NHWC store: one packed-f16 b128 per lane.
  const int coutBase = ct * 16 + half * 8;
  const float* bp = bias + coutBase;
  v8h ov;
  #pragma unroll
  for (int r = 0; r < 8; ++r) {
    float v = acc[r] + bp[r];
    if (RELU) v = fmaxf(v, 0.f);
    ov[r] = (_Float16)v;
  }
  _Float16* op = out + (((size_t)b * Hp + h + 1) * Wp + (w0 + lp + 1)) * COUT + coutBase;
  *(v8h*)op = ov;
}

// ---------------------------------------------------------------------------
// 2x2 / stride-2 max pool, padded NHWC f16 -> padded NHWC f16.
// ---------------------------------------------------------------------------
__global__ void maxpool2_kernel(const _Float16* __restrict__ in, _Float16* __restrict__ out,
                                int n, int Ho, int Wo, int C) {
  int i = blockIdx.x * 256 + threadIdx.x;
  if (i >= n) return;
  int c  = i % C;
  int t  = i / C;
  int wo = t % Wo;
  int ho = (t / Wo) % Ho;
  int b  = t / (Wo * Ho);
  const int Wi = 2 * Wo + 2, Hi = 2 * Ho + 2;
  const _Float16* p = in + ((size_t)b * Hi + 2 * ho + 1) * Wi * C + (size_t)(2 * wo + 1) * C + c;
  float m0 = fmaxf((float)p[0], (float)p[C]);
  float m1 = fmaxf((float)p[(size_t)Wi * C], (float)p[(size_t)Wi * C + C]);
  out[(((size_t)b * (Ho + 2) + ho + 1) * (Wo + 2) + wo + 1) * C + c] = (_Float16)fmaxf(m0, m1);
}

// ---------------------------------------------------------------------------
// 2x2 stride-2 transposed conv + bias + skip add (all padded NHWC f16).
// u[b,2h+i,2w+j,o] = bt[o] + d[b,2h+i,2w+j,o] + sum_ci e[b,h,w,ci]*wt[ci,o,i,j]
// ---------------------------------------------------------------------------
__global__ void convt_add_kernel(const _Float16* __restrict__ e,
                                 const float* __restrict__ wt,
                                 const float* __restrict__ bt,
                                 const _Float16* __restrict__ d,
                                 _Float16* __restrict__ u, int H, int W, int n) {
  int i = blockIdx.x * 256 + threadIdx.x;
  if (i >= n) return;
  const int W2 = 2 * W, H2 = 2 * H;
  int o  = i % 32;
  int t  = i / 32;
  int ow = t % W2;
  int oh = (t / W2) % H2;
  int b  = t / (W2 * H2);
  const int ii = oh & 1, jj = ow & 1;
  const int hh = oh >> 1, ww = ow >> 1;
  const size_t pidx = ((((size_t)b * (H2 + 2) + oh + 1) * (W2 + 2)) + ow + 1) * 32 + o;
  float s = bt[o] + (float)d[pidx];
  const _Float16* ep = e + (((size_t)b * (H + 2) + hh + 1) * (W + 2) + ww + 1) * 32;
  #pragma unroll
  for (int ci = 0; ci < 32; ++ci)
    s += (float)ep[ci] * wt[((ci * 32 + o) * 2 + ii) * 2 + jj];
  u[pidx] = (_Float16)s;
}

// ---------------------------------------------------------------------------
// 5x5 dilated (DIL=2) bilateral filter over 27 channels: 11 fp32 guide/value
// channels from x (NCHW) + 16 f16 feature channels (padded NHWC, contiguous
// per pixel).  Only the 11 output channels accumulate num; all 27 feed the
// range weight.  OOB taps skipped == reference's zero-padded mask.
// ---------------------------------------------------------------------------
__global__ __launch_bounds__(256)
void bilateral_kernel(const float* __restrict__ x, const _Float16* __restrict__ f,
                      const float* __restrict__ fp, float* __restrict__ out,
                      int H, int W) {
  int i = blockIdx.x * 256 + threadIdx.x;
  const int total = 4 * H * W;
  if (i >= total) return;
  const int w = i % W;
  const int h = (i / W) % H;
  const int b = i / (W * H);
  const size_t HWp = (size_t)H * W;
  const float* xb = x + (size_t)b * 16 * HWp;
  const _Float16* fb = f + (size_t)b * (H + 2) * (W + 2) * 16;
  const size_t p = (size_t)h * W + w;

  float cw[27], cen[27];
  #pragma unroll
  for (int c = 0; c < 27; ++c) cw[c] = fp[c];
  const float sy = fp[27], sx = fp[28];
  #pragma unroll
  for (int c = 0; c < 11; ++c) cen[c] = xb[c * HWp + p];
  {
    const _Float16* fc = fb + (((size_t)h + 1) * (W + 2) + w + 1) * 16;
    #pragma unroll
    for (int c = 0; c < 16; ++c) cen[11 + c] = (float)fc[c];
  }

  float num[11];
  #pragma unroll
  for (int c = 0; c < 11; ++c) num[c] = 0.f;
  float den = 0.f;

  #pragma unroll
  for (int dy = -2; dy <= 2; ++dy) {
    #pragma unroll
    for (int dx = -2; dx <= 2; ++dx) {
      const int oy = dy * 2, ox = dx * 2;
      const int yy = h + oy, xx = w + ox;
      if ((unsigned)yy >= (unsigned)H || (unsigned)xx >= (unsigned)W) continue;
      const size_t q = (size_t)yy * W + xx;
      float s[27];
      #pragma unroll
      for (int c = 0; c < 11; ++c) s[c] = xb[c * HWp + q];
      const _Float16* fq = fb + (((size_t)yy + 1) * (W + 2) + xx + 1) * 16;
      #pragma unroll
      for (int c = 0; c < 16; ++c) s[11 + c] = (float)fq[c];
      float lg = sy * (float)(oy * oy) + sx * (float)(ox * ox);
      #pragma unroll
      for (int c = 0; c < 27; ++c) {
        float d0 = s[c] - cen[c];
        lg = fmaf(cw[c], d0 * d0, lg);
      }
      const float wgt = __expf(lg);
      den += wgt;
      #pragma unroll
      for (int c = 0; c < 11; ++c) num[c] = fmaf(wgt, s[c], num[c]);
    }
  }
  const float inv = 1.f / den;
  #pragma unroll
  for (int c = 0; c < 11; ++c)
    out[((size_t)b * 11 + c) * HWp + p] = num[c] * inv;
}

// ---------------------------------------------------------------------------
// Host orchestration (graph-capture safe).  Input order:
// x, (w0,b0)...(w8,b8), wt, bt, fparams
// ---------------------------------------------------------------------------
extern "C" void kernel_launch(void* const* d_in, const int* in_sizes, int n_in,
                              void* d_out, int out_size, void* d_ws, size_t ws_size,
                              hipStream_t stream) {
  (void)in_sizes; (void)n_in; (void)out_size; (void)ws_size;
  const float* x = (const float*)d_in[0];
  const float* wl[9];
  const float* bl[9];
  for (int i = 0; i < 9; ++i) { wl[i] = (const float*)d_in[1 + 2 * i];
                                bl[i] = (const float*)d_in[2 + 2 * i]; }
  const float* wt = (const float*)d_in[19];
  const float* bt = (const float*)d_in[20];
  const float* fp = (const float*)d_in[21];
  float* out = (float*)d_out;

  // Padded NHWC f16 buffer sizes (bytes)
  const size_t P16  = 4ull * 514 * 514 * 16 * 2;   // ~33.8 MB
  const size_t P32  = 4ull * 514 * 514 * 32 * 2;   // ~67.6 MB
  const size_t P32h = 4ull * 258 * 258 * 32 * 2;   // ~17.0 MB

  char* ws = (char*)d_ws;
  size_t off = 0;
  auto alloc = [&](size_t sz) -> void* {
    void* pp = ws + off;
    off += (sz + 255) & ~(size_t)255;
    return pp;
  };
  _Float16* X16 = (_Float16*)alloc(P16);
  _Float16* tA  = (_Float16*)alloc(P16);
  _Float16* tB  = (_Float16*)alloc(P16);
  _Float16* D   = (_Float16*)alloc(P32);
  _Float16* E0  = (_Float16*)alloc(P32h);
  _Float16* E1  = (_Float16*)alloc(P32h);
  _Float16* U   = (_Float16*)alloc(P32);
  _Float16* WP  = (_Float16*)alloc(9u * 32768u);
  _Float16* F   = X16;   // X16 dead after conv0; reuse for final features

  // ---- weight packing ----
  const int cins[9]  = {16, 16, 16, 32, 32, 32, 32, 16, 16};
  const int couts[9] = {16, 16, 32, 32, 32, 32, 16, 16, 16};
  for (int i = 0; i < 9; ++i) {
    const int nch = (cins[i] == 32) ? 9 : 5;
    const int nt  = couts[i] / 16;
    pack_w_kernel<<<nt * nch, 32, 0, stream>>>(wl[i], WP + (size_t)i * 16384, cins[i]);
  }

  // ---- layout conversion + halo zeroing ----
  {
    const int n = 4 * 512 * 512 * 16;
    x_to_nhwc_kernel<<<(n + 255) / 256, 256, 0, stream>>>(x, X16);
    auto zb = [&](_Float16* buf, int H, int W, int C) {
      const int tot = 4 * (2 * (W + 2) + 2 * H) * C;
      zero_border_kernel<<<(tot + 255) / 256, 256, 0, stream>>>(buf, 4, H, W, C);
    };
    zb(X16, 512, 512, 16);
    zb(tA, 512, 512, 16);
    zb(tB, 512, 512, 16);
    zb(D, 512, 512, 32);
    zb(E0, 256, 256, 32);
    zb(E1, 256, 256, 32);
    zb(U, 512, 512, 32);
  }

  // ---- encoder front: 512x512 ----
  {
    dim3 g(4, 512, 4);
    conv3x3_wmma_kernel<16, 16, true ><<<g, 256, 0, stream>>>(X16, WP + 0 * 16384, bl[0], tA, 512, 512);
    conv3x3_wmma_kernel<16, 16, true ><<<g, 256, 0, stream>>>(tA,  WP + 1 * 16384, bl[1], tB, 512, 512);
    dim3 g2(4, 512, 8);
    conv3x3_wmma_kernel<16, 32, false><<<g2, 256, 0, stream>>>(tB, WP + 2 * 16384, bl[2], D, 512, 512);
  }

  // ---- maxpool + mid: 256x256 ----
  {
    const int n = 4 * 256 * 256 * 32;
    maxpool2_kernel<<<(n + 255) / 256, 256, 0, stream>>>(D, E0, n, 256, 256, 32);
    dim3 gh(2, 256, 8);
    conv3x3_wmma_kernel<32, 32, true><<<gh, 256, 0, stream>>>(E0, WP + 3 * 16384, bl[3], E1, 256, 256);
    conv3x3_wmma_kernel<32, 32, true><<<gh, 256, 0, stream>>>(E1, WP + 4 * 16384, bl[4], E0, 256, 256);
    conv3x3_wmma_kernel<32, 32, true><<<gh, 256, 0, stream>>>(E0, WP + 5 * 16384, bl[5], E1, 256, 256);
  }

  // ---- upsample + skip ----
  {
    const int n = 4 * 512 * 512 * 32;
    convt_add_kernel<<<(n + 255) / 256, 256, 0, stream>>>(E1, wt, bt, D, U, 256, 256, n);
  }

  // ---- decoder tail: 512x512 ----
  {
    dim3 g1(4, 512, 4);
    conv3x3_wmma_kernel<32, 16, true ><<<g1, 256, 0, stream>>>(U,  WP + 6 * 16384, bl[6], tA, 512, 512);
    conv3x3_wmma_kernel<16, 16, true ><<<g1, 256, 0, stream>>>(tA, WP + 7 * 16384, bl[7], tB, 512, 512);
    conv3x3_wmma_kernel<16, 16, false><<<g1, 256, 0, stream>>>(tB, WP + 8 * 16384, bl[8], F, 512, 512);
  }

  // ---- bilateral filter ----
  {
    const int n = 4 * 512 * 512;
    bilateral_kernel<<<(n + 255) / 256, 256, 0, stream>>>(x, F, fp, out, 512, 512);
  }
}
